// WAPDecoder_73547019976867
// MI455X (gfx1250) — matrix operand order repaired and checked by reading, not compiled, not run on hardware
//
#include <hip/hip_runtime.h>

typedef __attribute__((ext_vector_type(16))) __bf16 v16bf;
typedef __attribute__((ext_vector_type(8)))  float  v8f;

#define Bn   512
#define Ln   1024
#define Tn   128
#define NCn  101
#define En   64
#define EMBn 30
#define Hn   50
#define An   50
#define A1S  54          // padded a1 row stride (bf16 elements), dword multiple
#define VP   160         // padded vec width (K for logits GEMM)
#define NCP  112         // padded NC (7 n-tiles)

__device__ __forceinline__ float fast_tanh(float x) {
    x = fminf(fmaxf(x, -20.0f), 20.0f);
    float e = __expf(2.0f * x);
    return (e - 1.0f) / (e + 1.0f);
}
__device__ __forceinline__ float fast_sigmoid(float x) {
    return 1.0f / (1.0f + __expf(-x));
}
// bf16 -> f32 unpack from a packed dword: pure shift/mask, no cvt needed
__device__ __forceinline__ float bflo(unsigned int p) { return __uint_as_float(p << 16); }
__device__ __forceinline__ float bfhi(unsigned int p) { return __uint_as_float(p & 0xffff0000u); }

__device__ __forceinline__ v16bf pack16(float4 f0, float4 f1, float4 f2, float4 f3) {
    v16bf r;
    r[0]=(__bf16)f0.x; r[1]=(__bf16)f0.y; r[2]=(__bf16)f0.z; r[3]=(__bf16)f0.w;
    r[4]=(__bf16)f1.x; r[5]=(__bf16)f1.y; r[6]=(__bf16)f1.z; r[7]=(__bf16)f1.w;
    r[8]=(__bf16)f2.x; r[9]=(__bf16)f2.y; r[10]=(__bf16)f2.z; r[11]=(__bf16)f2.w;
    r[12]=(__bf16)f3.x; r[13]=(__bf16)f3.y; r[14]=(__bf16)f3.z; r[15]=(__bf16)f3.w;
    return r;
}

// ---------------- Kernel 0: pad W_fc [101,144] -> bf16 [112,160] ----------------
__global__ __launch_bounds__(256) void prep_wfc_kernel(const float* __restrict__ Wfc,
                                                       __bf16* __restrict__ out) {
    int i = blockIdx.x * 256 + threadIdx.x;
    if (i >= NCP * VP) return;
    int n = i / VP, k = i % VP;
    out[i] = (n < NCn && k < 144) ? (__bf16)Wfc[n * 144 + k] : (__bf16)0.0f;
}

// ---------------- Kernel 1: a1 = inputs_embeds @ W_att1^T  (WMMA bf16) ----------
// M = B*L = 524288, N = 50 (4 n-tiles of 16), K = 64 (2 k-chunks of 32)
__global__ __launch_bounds__(256, 1) void a1_kernel(const float* __restrict__ embeds,
                                                    const float* __restrict__ Watt1,
                                                    __bf16* __restrict__ a1ws) {
    __shared__ __align__(32) __bf16 sB[64 * 64];   // W_att1 padded [N=64][K=64]
    const int tid = threadIdx.x;
    for (int i = tid; i < 64 * 64; i += 256) {
        int n = i >> 6, k = i & 63;
        sB[i] = (n < An) ? (__bf16)Watt1[n * 64 + k] : (__bf16)0.0f;
    }
    __syncthreads();

    const int wave = tid >> 5, lane = tid & 31;
    const int m = lane & 15, grp = lane >> 4;
    const long mtile = (long)blockIdx.x * 8 + wave;
    const long rowBase = mtile * 16;

    const float* arow = embeds + (rowBase + m) * 64;
    v16bf a0, a1v;
    {   // k-chunk 0: K = grp*16+[0,16)
        float4 f0 = *(const float4*)(arow + grp * 16);
        float4 f1 = *(const float4*)(arow + grp * 16 + 4);
        float4 f2 = *(const float4*)(arow + grp * 16 + 8);
        float4 f3 = *(const float4*)(arow + grp * 16 + 12);
        a0 = pack16(f0, f1, f2, f3);
    }
    {   // k-chunk 1: +32
        float4 f0 = *(const float4*)(arow + 32 + grp * 16);
        float4 f1 = *(const float4*)(arow + 32 + grp * 16 + 4);
        float4 f2 = *(const float4*)(arow + 32 + grp * 16 + 8);
        float4 f3 = *(const float4*)(arow + 32 + grp * 16 + 12);
        a1v = pack16(f0, f1, f2, f3);
    }

    #pragma unroll
    for (int nt = 0; nt < 4; ++nt) {
        v8f acc = {0.0f,0.0f,0.0f,0.0f,0.0f,0.0f,0.0f,0.0f};
        v16bf b0 = *(const v16bf*)&sB[(nt * 16 + m) * 64 + grp * 16];
        v16bf b1 = *(const v16bf*)&sB[(nt * 16 + m) * 64 + 32 + grp * 16];
        acc = __builtin_amdgcn_wmma_f32_16x16x32_bf16(false, a0, false, b0, (short)0, acc, false, false);
        acc = __builtin_amdgcn_wmma_f32_16x16x32_bf16(false, a1v, false, b1, (short)0, acc, false, false);
        int col = nt * 16 + m;
        if (col < An) {
            #pragma unroll
            for (int v = 0; v < 8; ++v) {
                long row = rowBase + v + 8 * grp;
                a1ws[row * A1S + col] = (__bf16)acc[v];
            }
        }
    }
}

// ---------------- Kernel 2: recurrent attentive GRU, all state in LDS -----------
__global__ __launch_bounds__(256, 1) void decoder_kernel(
    const float* __restrict__ embeds,   // [B,L,E]
    const int*   __restrict__ ids,      // [B,T]
    const float* __restrict__ etab,     // [NC,EMB]
    const float* __restrict__ Watt2,    // [A,H]
    const float* __restrict__ Wproj,    // [A]
    const float* __restrict__ bproj,    // [1]
    const float* __restrict__ Wih,      // [150,94]
    const float* __restrict__ Whh,      // [150,50]
    const __bf16* __restrict__ a1ws,    // [B,L,A1S]
    __bf16* __restrict__ vecs)          // [B*T,VP]
{
    __shared__ __align__(16) __bf16 sA1[Ln * A1S];    // 110,592 B
    __shared__ __align__(16) __bf16 sEmb[Ln * En];    // 131,072 B
    __shared__ __align__(16) __bf16 sWih[150 * 94];   //  28,200 B
    __shared__ __align__(16) __bf16 sWhh[150 * 50];   //  15,000 B
    __shared__ float sWatt2[50 * 50];                 //  10,000 B
    __shared__ float sWp[50];
    __shared__ float sScores[Ln];
    __shared__ float sRedA[8], sRedB[8];
    __shared__ float sHW[50], sH[50], sHnew[50], sCtx[64], sCtxp[8 * 64];
    __shared__ float sEt[EMBn], sInp[94], sGi[150], sGh[150];

    const int tid = threadIdx.x;
    const int wave = tid >> 5, lane = tid & 31;
    const int b = blockIdx.x;

    {   // a1 tile -> LDS (bf16, dword copies)
        const unsigned int* src = (const unsigned int*)(a1ws + (size_t)b * Ln * A1S);
        unsigned int* dst = (unsigned int*)sA1;
        for (int i = tid; i < Ln * A1S / 2; i += 256) dst[i] = src[i];
    }
    {   // inputs_embeds tile -> LDS as bf16
        const float4* src = (const float4*)(embeds + (size_t)b * Ln * En);
        for (int i = tid; i < Ln * En / 4; i += 256) {
            float4 v = src[i];
            __bf16* d = &sEmb[i * 4];
            d[0] = (__bf16)v.x; d[1] = (__bf16)v.y; d[2] = (__bf16)v.z; d[3] = (__bf16)v.w;
        }
    }
    for (int i = tid; i < 150 * 94; i += 256) sWih[i] = (__bf16)Wih[i];
    for (int i = tid; i < 150 * 50; i += 256) sWhh[i] = (__bf16)Whh[i];
    for (int i = tid; i < 50 * 50;  i += 256) sWatt2[i] = Watt2[i];
    if (tid < 50) { sWp[tid] = Wproj[tid]; sH[tid] = 0.0f; }
    const float bp = bproj[0];
    __syncthreads();

    for (int t = 0; t < Tn; ++t) {
        // ---- stage 1: hW = W_att2 @ h ; fetch e_t in parallel
        if (tid < 50) {
            float acc = 0.0f;
            const float* wr = &sWatt2[tid * 50];
            #pragma unroll 10
            for (int j = 0; j < 50; ++j) acc += wr[j] * sH[j];
            sHW[tid] = acc;
        } else if (tid >= 64 && tid < 64 + EMBn) {
            int id = ids[b * Tn + t];
            id = id < 0 ? 0 : (id > NCn - 1 ? NCn - 1 : id);
            sEt[tid - 64] = etab[id * EMBn + (tid - 64)];
        }
        __syncthreads();                                   // (1)

        // ---- stage 2: scores[l] = W_proj . tanh(a1[l,:] + hW) + b_proj
        float sloc[4];
        float lmax = -3.4e38f;
        #pragma unroll
        for (int c = 0; c < 4; ++c) {
            int l = tid + 256 * c;
            const unsigned int* ar = (const unsigned int*)&sA1[l * A1S];
            float s = bp;
            #pragma unroll 5
            for (int a2 = 0; a2 < 25; ++a2) {
                unsigned int pk = ar[a2];
                s += sWp[2 * a2]     * fast_tanh(bflo(pk) + sHW[2 * a2]);
                s += sWp[2 * a2 + 1] * fast_tanh(bfhi(pk) + sHW[2 * a2 + 1]);
            }
            sloc[c] = s;
            lmax = fmaxf(lmax, s);
        }
        // wave32 shuffle reduction for max, then 8-slot combine
        #pragma unroll
        for (int off = 16; off > 0; off >>= 1)
            lmax = fmaxf(lmax, __shfl_xor(lmax, off));
        if (lane == 0) sRedA[wave] = lmax;
        __syncthreads();                                   // (2)
        float M = fmaxf(fmaxf(fmaxf(sRedA[0], sRedA[1]), fmaxf(sRedA[2], sRedA[3])),
                        fmaxf(fmaxf(sRedA[4], sRedA[5]), fmaxf(sRedA[6], sRedA[7])));

        float lsum = 0.0f;
        #pragma unroll
        for (int c = 0; c < 4; ++c) {
            float e = __expf(sloc[c] - M);
            sScores[tid + 256 * c] = e;
            lsum += e;
        }
        #pragma unroll
        for (int off = 16; off > 0; off >>= 1)
            lsum += __shfl_xor(lsum, off);
        if (lane == 0) sRedB[wave] = lsum;
        __syncthreads();                                   // (3)
        float inv = 1.0f / (sRedB[0] + sRedB[1] + sRedB[2] + sRedB[3] +
                            sRedB[4] + sRedB[5] + sRedB[6] + sRedB[7]);

        // ---- stage 3: context[e] = (sum_l alpha_l * emb[l,e]) * inv
        {   // 32 e-pairs x 8 l-chunks; one packed dword per row per thread
            int ep = tid & 31, c = tid >> 5;
            float acc0 = 0.0f, acc1 = 0.0f;
            const unsigned int* em = (const unsigned int*)sEmb;  // row = 32 dwords
            int base = c * 128;
            for (int l = base; l < base + 128; ++l) {
                float al = sScores[l];
                unsigned int pk = em[l * 32 + ep];
                acc0 += al * bflo(pk);
                acc1 += al * bfhi(pk);
            }
            sCtxp[c * 64 + 2 * ep]     = acc0;
            sCtxp[c * 64 + 2 * ep + 1] = acc1;
        }
        __syncthreads();                                   // (4)
        if (tid < 64) {
            float cx = 0.0f;
            #pragma unroll
            for (int c = 0; c < 8; ++c) cx += sCtxp[c * 64 + tid];
            cx *= inv;
            sCtx[tid] = cx;
            sInp[EMBn + tid] = cx;                          // inp[30..93] = context
        } else if (tid >= 64 && tid < 64 + EMBn) {
            sInp[tid - 64] = sEt[tid - 64];                 // inp[0..29] = e_t
        }
        __syncthreads();                                   // (5)

        // ---- stage 4: GRU gate GEMVs (packed bf16 dword reads)
        if (tid < 150) {
            float gi = 0.0f;
            const unsigned int* wr = (const unsigned int*)&sWih[tid * 94];
            for (int j = 0; j < 47; ++j) {
                unsigned int pk = wr[j];
                gi += bflo(pk) * sInp[2 * j] + bfhi(pk) * sInp[2 * j + 1];
            }
            float gh = 0.0f;
            const unsigned int* wr2 = (const unsigned int*)&sWhh[tid * 50];
            #pragma unroll 5
            for (int j = 0; j < 25; ++j) {
                unsigned int pk = wr2[j];
                gh += bflo(pk) * sH[2 * j] + bfhi(pk) * sH[2 * j + 1];
            }
            sGi[tid] = gi; sGh[tid] = gh;
        }
        __syncthreads();                                   // (6)
        if (tid < 50) {
            float r = fast_sigmoid(sGi[tid] + sGh[tid]);
            float z = fast_sigmoid(sGi[50 + tid] + sGh[50 + tid]);
            float n = fast_tanh(sGi[100 + tid] + r * sGh[100 + tid]);
            sHnew[tid] = (1.0f - z) * n + z * sH[tid];
        }
        __syncthreads();                                   // (7)
        if (tid < VP) {
            float v = tid < EMBn ? sEt[tid]
                    : (tid < 94 ? sCtx[tid - EMBn]
                    : (tid < 144 ? sHnew[tid - 94] : 0.0f));
            vecs[((size_t)b * Tn + t) * VP + tid] = (__bf16)v;
        }
        if (tid < 50) sH[tid] = sHnew[tid];
        __syncthreads();                                   // (8)
    }
}

// ---------------- Kernel 3: logits = vecs @ W_fc^T + b_fc  (WMMA bf16) ----------
// M = B*T = 65536, N = 112 (7 tiles, write 101), K = 160 (5 chunks)
__global__ __launch_bounds__(256, 1) void logits_kernel(const __bf16* __restrict__ vecs,
                                                        const __bf16* __restrict__ wfc,
                                                        const float* __restrict__ bfc,
                                                        float* __restrict__ out) {
    const int tid = threadIdx.x;
    const int wave = tid >> 5, lane = tid & 31;
    const int m = lane & 15, grp = lane >> 4;
    const long mtile = (long)blockIdx.x * 8 + wave;
    const long rowBase = mtile * 16;

    const __bf16* arow = vecs + (rowBase + m) * VP;
    v16bf a[5];
    #pragma unroll
    for (int kc = 0; kc < 5; ++kc) {
        uint4 lo = *(const uint4*)(arow + kc * 32 + grp * 16);
        uint4 hi = *(const uint4*)(arow + kc * 32 + grp * 16 + 8);
        union { v16bf v; uint4 u[2]; } tmp;
        tmp.u[0] = lo; tmp.u[1] = hi;
        a[kc] = tmp.v;
    }

    for (int nt = 0; nt < 7; ++nt) {
        v8f acc = {0.0f,0.0f,0.0f,0.0f,0.0f,0.0f,0.0f,0.0f};
        #pragma unroll
        for (int kc = 0; kc < 5; ++kc) {
            v16bf bv = *(const v16bf*)(wfc + (nt * 16 + m) * VP + kc * 32 + grp * 16);
            acc = __builtin_amdgcn_wmma_f32_16x16x32_bf16(false, a[kc], false, bv, (short)0, acc, false, false);
        }
        int col = nt * 16 + m;
        if (col < NCn) {
            float bias = bfc[col];
            #pragma unroll
            for (int v = 0; v < 8; ++v) {
                long row = rowBase + v + 8 * grp;
                out[row * NCn + col] = acc[v] + bias;
            }
        }
    }
}

extern "C" void kernel_launch(void* const* d_in, const int* in_sizes, int n_in,
                              void* d_out, int out_size, void* d_ws, size_t ws_size,
                              hipStream_t stream) {
    const float* embeds = (const float*)d_in[0];
    const int*   ids    = (const int*)d_in[1];
    const float* etab   = (const float*)d_in[2];
    const float* Watt1  = (const float*)d_in[3];
    const float* Watt2  = (const float*)d_in[4];
    const float* Wproj  = (const float*)d_in[5];
    const float* bproj  = (const float*)d_in[6];
    const float* Wih    = (const float*)d_in[7];
    const float* Whh    = (const float*)d_in[8];
    const float* Wfc    = (const float*)d_in[9];
    const float* bfc    = (const float*)d_in[10];
    float* out = (float*)d_out;

    const size_t a1_bytes  = (size_t)Bn * Ln * A1S * 2;      // 56.6 MB
    const size_t vec_bytes = (size_t)Bn * Tn * VP * 2;       // 21.0 MB
    const size_t wfc_bytes = (size_t)NCP * VP * 2;           // 35.8 KB
    if (ws_size < a1_bytes + vec_bytes + wfc_bytes) return;

    char* ws = (char*)d_ws;
    __bf16* a1ws   = (__bf16*)ws;
    __bf16* vecsws = (__bf16*)(ws + a1_bytes);
    __bf16* wfcws  = (__bf16*)(ws + a1_bytes + vec_bytes);

    prep_wfc_kernel<<<(NCP * VP + 255) / 256, 256, 0, stream>>>(Wfc, wfcws);
    a1_kernel<<<(Bn * Ln) / (16 * 8), 256, 0, stream>>>(embeds, Watt1, a1ws);
    decoder_kernel<<<Bn, 256, 0, stream>>>(embeds, ids, etab, Watt2, Wproj, bproj,
                                           Wih, Whh, a1ws, vecsws);
    logits_kernel<<<(Bn * Tn) / (16 * 8), 256, 0, stream>>>(vecsws, wfcws, bfc, out);
}
